// TemporalContrastiveLoss_46213848105954
// MI455X (gfx1250) — matrix operand order repaired and checked by reading, not compiled, not run on hardware
//
#include <hip/hip_runtime.h>
#include <hip/hip_bf16.h>

// ---------------------------------------------------------------------------
// TemporalContrastiveLoss on MI455X (gfx1250, wave32, WMMA)
//
// N = 16*512 = 8192 rows, D = 256.
// loss = -mean_i log( (pos_sum_i + 1e-8) / (sum_{j!=i} exp(sim_ij) + 1e-8) )
// sim = (normalize(E) @ normalize(E)^T) / 0.07
//
// Dominant cost: 8192^2 x 256 x 2 = 34.4 GFLOP GEMM over a 4MB (f16) matrix
// that lives entirely in L2 -> compute bound -> v_wmma_f32_16x16x32_f16.
// Only row sums of exp(sim) are needed, never the full matrix.
// ---------------------------------------------------------------------------

typedef __attribute__((ext_vector_type(16))) _Float16 v16h;
typedef __attribute__((ext_vector_type(8)))  _Float16 v8h;
typedef __attribute__((ext_vector_type(8)))  float    v8f;

#define NROWS   8192
#define SEQ     512
#define DIM     256
#define INVT    14.2857142857142857f   /* 1/0.07 */
#define NTILES  (NROWS / 16)           /* 512 i/j tiles */
#define KCHUNKS (DIM / 32)             /* 8 WMMA K-steps of 32 */
#define JCHUNKS 8
#define JT_PER_CHUNK (NTILES / JCHUNKS) /* 64 */

// ---------------------------------------------------------------------------
// Kernel A: L2-normalize each row (fp32), convert to f16, store inv_norm,
// self-term exp(<h,h>/T) computed from the *converted* f16 values, zero S_all.
// One wave per row; lane handles 8 consecutive elements.
// ---------------------------------------------------------------------------
__global__ __launch_bounds__(256) void tcl_normalize_kernel(
    const float* __restrict__ emb, _Float16* __restrict__ Ef,
    float* __restrict__ inv_norm, float* __restrict__ S_self,
    float* __restrict__ S_all) {
  const int lane = threadIdx.x & 31;
  const int wave = threadIdx.x >> 5;
  const int row  = blockIdx.x * 8 + wave;

  const float4* p = (const float4*)(emb + (size_t)row * DIM + lane * 8);
  float4 x0 = p[0], x1 = p[1];

  float ss = x0.x*x0.x + x0.y*x0.y + x0.z*x0.z + x0.w*x0.w
           + x1.x*x1.x + x1.y*x1.y + x1.z*x1.z + x1.w*x1.w;
#pragma unroll
  for (int off = 16; off >= 1; off >>= 1) ss += __shfl_xor(ss, off, 32);

  const float scale = 1.0f / fmaxf(sqrtf(ss), 1e-12f);

  v8h h;
  h[0] = (_Float16)(x0.x * scale); h[1] = (_Float16)(x0.y * scale);
  h[2] = (_Float16)(x0.z * scale); h[3] = (_Float16)(x0.w * scale);
  h[4] = (_Float16)(x1.x * scale); h[5] = (_Float16)(x1.y * scale);
  h[6] = (_Float16)(x1.z * scale); h[7] = (_Float16)(x1.w * scale);
  *(v8h*)(Ef + (size_t)row * DIM + lane * 8) = h;

  // self dot in the same precision the WMMA sees
  float sq = 0.f;
#pragma unroll
  for (int t = 0; t < 8; ++t) { float f = (float)h[t]; sq += f * f; }
#pragma unroll
  for (int off = 16; off >= 1; off >>= 1) sq += __shfl_xor(sq, off, 32);

  if (lane == 0) {
    inv_norm[row] = scale;
    S_self[row]   = __expf(sq * INVT);
    S_all[row]    = 0.0f;               // init for the atomic accumulation
  }
}

// ---------------------------------------------------------------------------
// Kernel B: exact fp32 positive sum. For each row i, <=10 neighbors j with
// same batch and 0 < |pos_i - pos_j| <= 5. One wave per row.
// ---------------------------------------------------------------------------
__global__ __launch_bounds__(256) void tcl_pos_kernel(
    const float* __restrict__ emb, const float* __restrict__ inv_norm,
    float* __restrict__ S_pos) {
  const int lane = threadIdx.x & 31;
  const int wave = threadIdx.x >> 5;
  const int i    = blockIdx.x * 8 + wave;
  const int p    = i & (SEQ - 1);

  const float4* pi = (const float4*)(emb + (size_t)i * DIM + lane * 8);
  float4 a0 = pi[0], a1 = pi[1];
  const float inv_i = inv_norm[i];

  float pos = 0.f;
#pragma unroll
  for (int d = -5; d <= 5; ++d) {
    if (d == 0) continue;
    const int pp = p + d;
    if (pp < 0 || pp >= SEQ) continue;
    const int j = i + d;
    const float4* pj = (const float4*)(emb + (size_t)j * DIM + lane * 8);
    float4 b0 = pj[0], b1 = pj[1];
    float dt = a0.x*b0.x + a0.y*b0.y + a0.z*b0.z + a0.w*b0.w
             + a1.x*b1.x + a1.y*b1.y + a1.z*b1.z + a1.w*b1.w;
#pragma unroll
    for (int off = 16; off >= 1; off >>= 1) dt += __shfl_xor(dt, off, 32);
    pos += __expf(dt * inv_i * inv_norm[j] * INVT);
  }
  if (lane == 0) S_pos[i] = pos;
}

// ---------------------------------------------------------------------------
// Kernel C: the big GEMM. Computes sim^T tiles (A = 16 rows j, B = 16 rows i)
// with v_wmma_f32_16x16x32_f16 and accumulates sum_j exp(sim_ij * 1/T) into
// S_all[i]. Per wave: fixed i-tile (B preloaded for all 8 K-chunks), loop
// over 64 j-tiles of this block's j-chunk.
//
// Fragment layouts as loaded here (wave32; A/B use contiguous k, not the ISA table order):
//   B (32x16 K x N): lane n<16 holds K=0..15 of col n, lane n+16 holds
//     K=16..31 -> one contiguous 16-element row chunk per lane.
//   A (16x32 M x K): lane l<16 holds K 0..15 of row l; lane l+16
//     holds K 16..31 -> two adjacent 8-element loads, same k order as B.
//   C/D (16x16): VGPR v = M row (j), lane = N col (i); lanes 16..31 are
//     M = v+8 -> sum-over-j is 8 regs + shfl_xor(16), then atomicAdd per i.
// ---------------------------------------------------------------------------
__global__ __launch_bounds__(256) void tcl_simrowsum_kernel(
    const _Float16* __restrict__ Ef, float* __restrict__ S_all) {
  const int lane  = threadIdx.x & 31;
  const int wave  = threadIdx.x >> 5;
  const int l16   = lane & 15;
  const int half  = lane >> 4;
  const int itile = blockIdx.x * 8 + wave;
  const int ibase = itile * 16;

  // Preload B fragments for this i-tile: 8 chunks x v16h = 64 VGPRs.
  v16h bfrag[KCHUNKS];
  const _Float16* brow = Ef + (size_t)(ibase + l16) * DIM + half * 16;
#pragma unroll
  for (int kc = 0; kc < KCHUNKS; ++kc)
    bfrag[kc] = *(const v16h*)(brow + kc * 32);

  const int jt0 = blockIdx.y * JT_PER_CHUNK;
  float rsum = 0.f;

  for (int jt = jt0; jt < jt0 + JT_PER_CHUNK; ++jt) {
    const int jbase = jt * 16;
    const _Float16* arow = Ef + (size_t)(jbase + l16) * DIM + half * 16;

    // Prefetch next j-tile's row (L2-resident; emits global_prefetch_b8).
    __builtin_prefetch(arow + DIM * 16, 0, 0);

    v8f acc = {};
#pragma unroll
    for (int kc = 0; kc < KCHUNKS; ++kc) {
      const _Float16* pa = arow + kc * 32;
      v8h a0 = *(const v8h*)(pa);        // K chunk at half*16
      v8h a1 = *(const v8h*)(pa + 8);    // K chunk at half*16 + 8
      v16h a = __builtin_shufflevector(a0, a1,
               0, 1, 2, 3, 4, 5, 6, 7, 8, 9, 10, 11, 12, 13, 14, 15);
      acc = __builtin_amdgcn_wmma_f32_16x16x32_f16(
          /*neg_a=*/false, a, /*neg_b=*/false, bfrag[kc],
          /*c_mod=*/(short)0, acc, /*reuse_a=*/false, /*reuse_b=*/false);
    }
    // exp over the 8 j's this lane holds (8 accumulator regs), sum over j.
#pragma unroll
    for (int e = 0; e < 8; ++e)
      rsum += __expf(acc[e] * INVT);
  }

  // lanes l and l+16 hold the two M-halves of the same column i = ibase+l16
  rsum += __shfl_xor(rsum, 16, 32);
  if (lane < 16) atomicAdd(&S_all[ibase + lane], rsum);
}

// ---------------------------------------------------------------------------
// Kernel D: loss = -mean_i log((S_pos+1e-8) / (S_all - S_self + 1e-8))
// ---------------------------------------------------------------------------
__global__ __launch_bounds__(256) void tcl_finalize_kernel(
    const float* __restrict__ S_pos, const float* __restrict__ S_all,
    const float* __restrict__ S_self, float* __restrict__ out) {
  __shared__ float sred[256];
  const int tid = threadIdx.x;
  float acc = 0.f;
  for (int i = tid; i < NROWS; i += 256) {
    const float num = S_pos[i] + 1e-8f;
    const float den = S_all[i] - S_self[i] + 1e-8f;
    acc += __logf(num / den);
  }
  sred[tid] = acc;
  __syncthreads();
#pragma unroll
  for (int s = 128; s >= 1; s >>= 1) {
    if (tid < s) sred[tid] += sred[tid + s];
    __syncthreads();
  }
  if (tid == 0) out[0] = -sred[0] / (float)NROWS;
}

// ---------------------------------------------------------------------------
extern "C" void kernel_launch(void* const* d_in, const int* in_sizes, int n_in,
                              void* d_out, int out_size, void* d_ws, size_t ws_size,
                              hipStream_t stream) {
  (void)in_sizes; (void)n_in; (void)out_size; (void)ws_size;
  const float* emb = (const float*)d_in[0];
  float* out = (float*)d_out;

  // Workspace layout
  char* base = (char*)d_ws;
  _Float16* Ef      = (_Float16*)base;                       // 8192*256*2 = 4 MB
  size_t off = (size_t)NROWS * DIM * sizeof(_Float16);
  float* S_all    = (float*)(base + off); off += NROWS * sizeof(float);
  float* S_pos    = (float*)(base + off); off += NROWS * sizeof(float);
  float* S_self   = (float*)(base + off); off += NROWS * sizeof(float);
  float* inv_norm = (float*)(base + off); off += NROWS * sizeof(float);

  // A: normalize + f16 convert + self-term + S_all init   (1024 blocks x 8 waves)
  tcl_normalize_kernel<<<NROWS / 8, 256, 0, stream>>>(emb, Ef, inv_norm, S_self, S_all);
  // B: exact fp32 positive sums
  tcl_pos_kernel<<<NROWS / 8, 256, 0, stream>>>(emb, inv_norm, S_pos);
  // C: WMMA GEMM row-sums of exp(sim): 512 i-tiles x 8 j-chunks
  dim3 gridC(NTILES / 8, JCHUNKS);
  tcl_simrowsum_kernel<<<gridC, 256, 0, stream>>>(Ef, S_all);
  // D: final reduction to scalar loss
  tcl_finalize_kernel<<<1, 256, 0, stream>>>(S_pos, S_all, S_self, out);
}
